// TransformerBlock_240518168597
// MI455X (gfx1250) — matrix operand order repaired and checked by reading, not compiled, not run on hardware
//
#include <hip/hip_runtime.h>
#include <cstddef>

// ---------- types ----------
typedef __attribute__((ext_vector_type(8)))  __bf16 v8bf;
typedef __attribute__((ext_vector_type(16))) __bf16 v16bf;
typedef __attribute__((ext_vector_type(8)))  float  v8f;
typedef __attribute__((ext_vector_type(4)))  int    v4i;

typedef __attribute__((address_space(1))) v4i* gptr_v4i;
typedef __attribute__((address_space(3))) v4i* lptr_v4i;

// ---------- gfx1250 async global->LDS path (ASYNCcnt) ----------
#if defined(__HIP_DEVICE_COMPILE__)
  #if __has_builtin(__builtin_amdgcn_global_load_async_to_lds_b128)
    #define USE_ASYNC_LDS 1
  #else
    #define USE_ASYNC_LDS 0
  #endif
#else
  #define USE_ASYNC_LDS 0
#endif

__device__ __forceinline__ v16bf cat8(v8bf lo, v8bf hi) {
    return __builtin_shufflevector(lo, hi, 0,1,2,3,4,5,6,7,8,9,10,11,12,13,14,15);
}

__device__ __forceinline__ __bf16 to_bf16(float f) {
    union { float f; unsigned u; } v; v.f = f;
    unsigned r = v.u + 0x7FFFu + ((v.u >> 16) & 1u);   // round-to-nearest-even
    unsigned short h = (unsigned short)(r >> 16);
    return __builtin_bit_cast(__bf16, h);
}

__device__ __forceinline__ v8f wmma_bf16(v16bf a, v16bf b, v8f c) {
    return __builtin_amdgcn_wmma_f32_16x16x32_bf16(
        false, a, false, b, (short)0, c, false, false);
}

// copy 16B global -> LDS (async DMA if available, else direct)
__device__ __forceinline__ void async_copy16(__bf16* dst_lds, const __bf16* src) {
#if USE_ASYNC_LDS
    v4i* s_gen = (v4i*)src;        // drop const, retype in generic AS
    v4i* d_gen = (v4i*)dst_lds;
    __builtin_amdgcn_global_load_async_to_lds_b128(
        (gptr_v4i)s_gen, (lptr_v4i)d_gen, 0, 0);
#else
    *(v8bf*)dst_lds = *(const v8bf*)src;
#endif
}

template <int N>
__device__ __forceinline__ void wait_async_le() {
#if USE_ASYNC_LDS
  #if __has_builtin(__builtin_amdgcn_s_wait_asynccnt)
    __builtin_amdgcn_s_wait_asynccnt(N);
  #else
    asm volatile("s_wait_asynccnt %0" :: "n"(N) : "memory");
  #endif
#endif
}

// ---------- weight cast + transpose: W[K][N] f32 -> Wt[N][K] bf16 ----------
__global__ __launch_bounds__(256)
void cast_transpose_kernel(const float* __restrict__ W, __bf16* __restrict__ Wt,
                           int K, int N) {
    size_t total = (size_t)K * (size_t)N;
    size_t stride = (size_t)gridDim.x * 256;
    for (size_t i = (size_t)blockIdx.x * 256 + threadIdx.x; i < total; i += stride) {
        size_t n = i / (size_t)K;
        size_t k = i % (size_t)K;
        Wt[i] = to_bf16(W[k * (size_t)N + n]);
    }
}

__global__ __launch_bounds__(256)
void pack_qkv_bias_kernel(const float* __restrict__ bq, const float* __restrict__ bk,
                          const float* __restrict__ bv, float* __restrict__ o, int D) {
    int i = blockIdx.x * 256 + threadIdx.x;
    if (i < D) { o[i] = bq[i]; o[D + i] = bk[i]; o[2 * D + i] = bv[i]; }
}

// ---------- LayerNorm: f32 row -> bf16 row ----------
__global__ __launch_bounds__(256)
void layernorm_bf16_kernel(const float* __restrict__ x, const float* __restrict__ g,
                           const float* __restrict__ b, __bf16* __restrict__ out, int C) {
    const int row = blockIdx.x;
    const float* xr = x + (size_t)row * C;
    float s = 0.f, ss = 0.f;
    for (int c = threadIdx.x; c < C; c += 256) { float v = xr[c]; s += v; ss += v * v; }
    #pragma unroll
    for (int m = 16; m >= 1; m >>= 1) { s += __shfl_xor(s, m); ss += __shfl_xor(ss, m); }
    __shared__ float sbuf[8], ssbuf[8];
    __shared__ float mu_sh, rstd_sh;
    if ((threadIdx.x & 31) == 0) { sbuf[threadIdx.x >> 5] = s; ssbuf[threadIdx.x >> 5] = ss; }
    __syncthreads();
    if (threadIdx.x == 0) {
        float S = 0.f, SS = 0.f;
        #pragma unroll
        for (int i = 0; i < 8; ++i) { S += sbuf[i]; SS += ssbuf[i]; }
        float mu = S / (float)C;
        float var = SS / (float)C - mu * mu;
        mu_sh = mu; rstd_sh = rsqrtf(var + 1e-5f);
    }
    __syncthreads();
    float mu = mu_sh, rstd = rstd_sh;
    for (int c = threadIdx.x; c < C; c += 256)
        out[(size_t)row * C + c] = to_bf16((xr[c] - mu) * rstd * g[c] + b[c]);
}

// ---------- WMMA GEMM: C[M,N] = A[M,K](bf16) * Bt[N,K](bf16) + bias (+resid) ----------
// Block tile 128x128, 8 waves (4 along M x 2 along N), wave tile 32x64.
// K-step 64, double-buffered LDS, async DMA issue-ahead of one tile.
__global__ __launch_bounds__(256)
void gemm_bf16_wmma_kernel(const __bf16* __restrict__ A, const __bf16* __restrict__ Bt,
                           const float* __restrict__ bias, const float* __restrict__ resid,
                           float* __restrict__ outF, __bf16* __restrict__ outB,
                           int M, int N, int K, int relu) {
    __shared__ __bf16 As[2][128][64];
    __shared__ __bf16 Bs[2][128][64];

    const int tid   = threadIdx.x;
    const int lane  = tid & 31;
    const int wave  = tid >> 5;
    const int waveM = wave & 3;      // 0..3 -> 32 rows each
    const int waveN = wave >> 2;     // 0..1 -> 64 cols each
    const int l15   = lane & 15;
    const int hi    = (lane >> 4) & 1;
    const int bm    = blockIdx.y * 128;
    const int bn    = blockIdx.x * 128;

    v8f acc[2][4];
    #pragma unroll
    for (int i = 0; i < 2; ++i)
        #pragma unroll
        for (int j = 0; j < 4; ++j)
            #pragma unroll
            for (int e = 0; e < 8; ++e) acc[i][j][e] = 0.0f;

    const int lr = tid >> 1;          // 0..127 : tile row for loads
    const int lc = (tid & 1) * 32;    // 0 or 32

    auto issue_tile = [&](int k0, int buf) {
        const __bf16* srcA = A  + (size_t)(bm + lr) * K + k0 + lc;
        const __bf16* srcB = Bt + (size_t)(bn + lr) * K + k0 + lc;
        __builtin_prefetch(srcA + 64, 0, 1);          // gfx1250 global_prefetch_b8
        __builtin_prefetch(srcB + 64, 0, 1);
#if USE_ASYNC_LDS
        #pragma unroll
        for (int j = 0; j < 4; ++j) async_copy16(&As[buf][lr][lc + j * 8], srcA + j * 8);
        #pragma unroll
        for (int j = 0; j < 4; ++j) async_copy16(&Bs[buf][lr][lc + j * 8], srcB + j * 8);
#else
        v8bf ta[4], tb[4];
        #pragma unroll
        for (int j = 0; j < 4; ++j) ta[j] = *(const v8bf*)(srcA + j * 8);
        #pragma unroll
        for (int j = 0; j < 4; ++j) tb[j] = *(const v8bf*)(srcB + j * 8);
        #pragma unroll
        for (int j = 0; j < 4; ++j) *(v8bf*)&As[buf][lr][lc + j * 8] = ta[j];
        #pragma unroll
        for (int j = 0; j < 4; ++j) *(v8bf*)&Bs[buf][lr][lc + j * 8] = tb[j];
#endif
    };

    auto compute_tile = [&](int buf) {
        #pragma unroll
        for (int s = 0; s < 2; ++s) {
            v16bf afrag[2];
            #pragma unroll
            for (int mt = 0; mt < 2; ++mt) {
                int row = waveM * 32 + mt * 16 + l15;
                int kb = s * 32 + hi * 8;
                v8bf lo = *(const v8bf*)&As[buf][row][kb];
                v8bf h8 = *(const v8bf*)&As[buf][row][kb + 16];
                afrag[mt] = cat8(lo, h8);
            }
            #pragma unroll
            for (int nt = 0; nt < 4; ++nt) {
                int col = waveN * 64 + nt * 16 + l15;
                int kb = s * 32 + hi * 8;
                v8bf lo = *(const v8bf*)&Bs[buf][col][kb];
                v8bf h8 = *(const v8bf*)&Bs[buf][col][kb + 16];
                v16bf bfrag = cat8(lo, h8);
                #pragma unroll
                for (int mt = 0; mt < 2; ++mt)
                    acc[mt][nt] = wmma_bf16(afrag[mt], bfrag, acc[mt][nt]);
            }
        }
    };

    const int ntiles = K >> 6;           // K / 64, K is a multiple of 64
    issue_tile(0, 0);
    for (int t = 0; t < ntiles; ++t) {
        const int buf = t & 1;
        if (t + 1 < ntiles) {
            issue_tile((t + 1) << 6, buf ^ 1);
            wait_async_le<8>();          // tile t done (8 newer ops may remain)
        } else {
            wait_async_le<0>();
        }
        __syncthreads();                 // tile t visible to all waves
        compute_tile(buf);
        __syncthreads();                 // all waves done with buf before overwrite
    }

    // epilogue
    #pragma unroll
    for (int mt = 0; mt < 2; ++mt)
        #pragma unroll
        for (int nt = 0; nt < 4; ++nt) {
            int col = bn + waveN * 64 + nt * 16 + l15;
            #pragma unroll
            for (int r = 0; r < 8; ++r) {
                int row = bm + waveM * 32 + mt * 16 + r + hi * 8;
                float v = acc[mt][nt][r];
                if (bias)  v += bias[col];
                if (relu)  v = fmaxf(v, 0.0f);
                if (resid) v += resid[(size_t)row * N + col];
                if (outF)  outF[(size_t)row * N + col] = v;
                if (outB)  outB[(size_t)row * N + col] = to_bf16(v);
            }
        }
}

// ---------- Flash attention (H=16, Dh=64), all matmuls via WMMA ----------
// grid: (T/128, B*16). Block 256 = 8 waves; wave owns 16 query rows.
__global__ __launch_bounds__(256)
void attention_wmma_kernel(const __bf16* __restrict__ qkv, // [B*T][3072] = q|k|v
                           __bf16* __restrict__ out,       // [B*T][1024]
                           int T) {
    __shared__ __bf16 Ks [128][64];    // K tile  [key][dh]
    __shared__ __bf16 Vts[64][128];    // V tile transposed [dh][key]
    __shared__ __bf16 Ps [8][16][128]; // per-wave P tile

    const int tid  = threadIdx.x;
    const int lane = tid & 31;
    const int wave = tid >> 5;
    const int l15  = lane & 15;
    const int hi   = (lane >> 4) & 1;
    const int bh   = blockIdx.y;
    const int b    = bh >> 4, h = bh & 15;
    const int q0   = blockIdx.x * 128;
    const size_t rs = 3072;

    const __bf16* qBase = qkv + (size_t)b * T * rs + h * 64;
    const __bf16* kBase = qBase + 1024;
    const __bf16* vBase = qBase + 2048;

    // Q fragments for this wave's 16 rows (dh split into two k-steps of 32)
    v16bf aq[2];
    {
        const __bf16* qp = qBase + (size_t)(q0 + wave * 16 + l15) * rs;
        #pragma unroll
        for (int s = 0; s < 2; ++s) {
            int kb = s * 32 + hi * 8;
            v8bf lo = *(const v8bf*)(qp + kb);
            v8bf h8 = *(const v8bf*)(qp + kb + 16);
            aq[s] = cat8(lo, h8);
        }
    }

    v8f o[4];
    float mrow[8], lrow[8];
    #pragma unroll
    for (int t = 0; t < 4; ++t)
        #pragma unroll
        for (int e = 0; e < 8; ++e) o[t][e] = 0.0f;
    #pragma unroll
    for (int r = 0; r < 8; ++r) { mrow[r] = -1e30f; lrow[r] = 0.0f; }

    for (int kt = 0; kt < T; kt += 128) {
        __syncthreads();
        {   // K tile via async DMA: 128x64, each thread 32 elems
            int r = tid >> 1, c = (tid & 1) * 32;
            const __bf16* src = kBase + (size_t)(kt + r) * rs + c;
            #pragma unroll
            for (int j = 0; j < 4; ++j)
                async_copy16(&Ks[r][c + j * 8], src + j * 8);
        }
        {   // V tile transposed (element-wise; overlaps the async K DMA)
            int key = tid >> 1, c = (tid & 1) * 32;
            const __bf16* src = vBase + (size_t)(kt + key) * rs + c;
            #pragma unroll
            for (int j = 0; j < 4; ++j) {
                v8bf v = *(const v8bf*)(src + j * 8);
                #pragma unroll
                for (int e = 0; e < 8; ++e) Vts[c + j * 8 + e][key] = v[e];
            }
        }
        wait_async_le<0>();
        __syncthreads();

        // S = Q * K^T  (16 x 128), eight 16x16 tiles, two k-steps over dh
        v8f s[8];
        #pragma unroll
        for (int nt = 0; nt < 8; ++nt) {
            #pragma unroll
            for (int e = 0; e < 8; ++e) s[nt][e] = 0.0f;
            int col = nt * 16 + l15;
            #pragma unroll
            for (int ss = 0; ss < 2; ++ss) {
                int kb = ss * 32 + hi * 8;
                v8bf lo = *(const v8bf*)&Ks[col][kb];
                v8bf h8 = *(const v8bf*)&Ks[col][kb + 16];
                s[nt] = wmma_bf16(aq[ss], cat8(lo, h8), s[nt]);
            }
        }

        // online softmax update (row = r + hi*8)
        float alpha[8];
        #pragma unroll
        for (int r = 0; r < 8; ++r) {
            float vm = -1e30f;
            #pragma unroll
            for (int nt = 0; nt < 8; ++nt) { s[nt][r] *= 0.125f; vm = fmaxf(vm, s[nt][r]); }
            #pragma unroll
            for (int m = 8; m >= 1; m >>= 1) vm = fmaxf(vm, __shfl_xor(vm, m));
            float mnew = fmaxf(mrow[r], vm);
            alpha[r] = __expf(mrow[r] - mnew);
            float ls = 0.f;
            #pragma unroll
            for (int nt = 0; nt < 8; ++nt) { float p = __expf(s[nt][r] - mnew); s[nt][r] = p; ls += p; }
            #pragma unroll
            for (int m = 8; m >= 1; m >>= 1) ls += __shfl_xor(ls, m);
            lrow[r] = lrow[r] * alpha[r] + ls;
            mrow[r] = mnew;
        }

        // spill P to LDS (C layout -> memory), rescale O
        #pragma unroll
        for (int nt = 0; nt < 8; ++nt)
            #pragma unroll
            for (int r = 0; r < 8; ++r)
                Ps[wave][r + hi * 8][nt * 16 + l15] = to_bf16(s[nt][r]);
        #pragma unroll
        for (int t = 0; t < 4; ++t)
            #pragma unroll
            for (int r = 0; r < 8; ++r) o[t][r] *= alpha[r];

        // O += P * V^T : contraction over 128 keys, four k-steps of 32
        #pragma unroll
        for (int ks = 0; ks < 4; ++ks) {
            int kb = ks * 32 + hi * 8;
            v8bf lo = *(const v8bf*)&Ps[wave][l15][kb];
            v8bf h8 = *(const v8bf*)&Ps[wave][l15][kb + 16];
            v16bf af = cat8(lo, h8);
            #pragma unroll
            for (int t = 0; t < 4; ++t) {
                int dh = t * 16 + l15;
                int kk = ks * 32 + hi * 8;
                v8bf blo = *(const v8bf*)&Vts[dh][kk];
                v8bf bhi = *(const v8bf*)&Vts[dh][kk + 16];
                o[t] = wmma_bf16(af, cat8(blo, bhi), o[t]);
            }
        }
    }

    // finalize: O /= l, store [b,t,h*64+dh] as bf16
    #pragma unroll
    for (int t = 0; t < 4; ++t)
        #pragma unroll
        for (int r = 0; r < 8; ++r) {
            int row = b * T + q0 + wave * 16 + r + hi * 8;
            int col = h * 64 + t * 16 + l15;
            out[(size_t)row * 1024 + col] = to_bf16(o[t][r] / lrow[r]);
        }
}

// ---------- host launcher ----------
extern "C" void kernel_launch(void* const* d_in, const int* in_sizes, int n_in,
                              void* d_out, int out_size, void* d_ws, size_t ws_size,
                              hipStream_t stream) {
    (void)in_sizes; (void)n_in; (void)out_size; (void)ws_size;
    const float* x    = (const float*)d_in[0];
    const float* ln1g = (const float*)d_in[1];
    const float* ln1b = (const float*)d_in[2];
    const float* ln2g = (const float*)d_in[3];
    const float* ln2b = (const float*)d_in[4];
    const float* Wq   = (const float*)d_in[5];
    const float* bq   = (const float*)d_in[6];
    const float* Wk   = (const float*)d_in[7];
    const float* bk   = (const float*)d_in[8];
    const float* Wv   = (const float*)d_in[9];
    const float* bv   = (const float*)d_in[10];
    const float* Wo   = (const float*)d_in[11];
    const float* bo   = (const float*)d_in[12];
    const float* W1   = (const float*)d_in[13];
    const float* b1   = (const float*)d_in[14];
    const float* W2   = (const float*)d_in[15];
    const float* b2   = (const float*)d_in[16];
    float* out = (float*)d_out;

    const int Bb = 2, T = 2048, D = 1024, HID = 4096;
    const int M = Bb * T, H3 = 3 * D;

    char* ws = (char*)d_ws;
    size_t off = 0;
    auto carve = [&](size_t bytes) -> char* {
        char* p = ws + off;
        off += (bytes + 255) & ~(size_t)255;
        return p;
    };
    __bf16* xn    = (__bf16*)carve((size_t)M * D * 2);
    __bf16* WqkvT = (__bf16*)carve((size_t)H3 * D * 2);
    float*  qkvb  = (float*) carve((size_t)H3 * 4);
    __bf16* WoT   = (__bf16*)carve((size_t)D * D * 2);
    __bf16* W1T   = (__bf16*)carve((size_t)HID * D * 2);
    __bf16* W2T   = (__bf16*)carve((size_t)D * HID * 2);
    __bf16* qkv   = (__bf16*)carve((size_t)M * H3 * 2);
    float*  x1    = (float*) carve((size_t)M * D * 4);
    __bf16* hbuf  = (__bf16*)carve((size_t)M * HID * 2);
    __bf16* attn  = xn;   // xn dead after QKV GEMM
    __bf16* xn2   = qkv;  // qkv dead after attention

    // weights -> bf16, transposed to [N][K]
    cast_transpose_kernel<<<512, 256, 0, stream>>>(Wq, WqkvT,                     D, D);
    cast_transpose_kernel<<<512, 256, 0, stream>>>(Wk, WqkvT + (size_t)D * D,     D, D);
    cast_transpose_kernel<<<512, 256, 0, stream>>>(Wv, WqkvT + (size_t)2 * D * D, D, D);
    cast_transpose_kernel<<<512, 256, 0, stream>>>(Wo, WoT,                       D, D);
    cast_transpose_kernel<<<1024, 256, 0, stream>>>(W1, W1T, D, HID);
    cast_transpose_kernel<<<1024, 256, 0, stream>>>(W2, W2T, HID, D);
    pack_qkv_bias_kernel<<<4, 256, 0, stream>>>(bq, bk, bv, qkvb, D);

    // LN1 -> fused QKV GEMM -> flash attention -> out-proj (+residual)
    layernorm_bf16_kernel<<<M, 256, 0, stream>>>(x, ln1g, ln1b, xn, D);
    gemm_bf16_wmma_kernel<<<dim3(H3 / 128, M / 128), 256, 0, stream>>>(
        xn, WqkvT, qkvb, nullptr, nullptr, qkv, M, H3, D, 0);
    attention_wmma_kernel<<<dim3(T / 128, Bb * 16), 256, 0, stream>>>(qkv, attn, T);
    gemm_bf16_wmma_kernel<<<dim3(D / 128, M / 128), 256, 0, stream>>>(
        attn, WoT, bo, x, x1, nullptr, M, D, D, 0);

    // LN2 -> MLP (relu) -> down-proj (+residual) -> d_out
    layernorm_bf16_kernel<<<M, 256, 0, stream>>>(x1, ln2g, ln2b, xn2, D);
    gemm_bf16_wmma_kernel<<<dim3(HID / 128, M / 128), 256, 0, stream>>>(
        xn2, W1T, b1, nullptr, nullptr, hbuf, M, HID, D, 1);
    gemm_bf16_wmma_kernel<<<dim3(D / 128, M / 128), 256, 0, stream>>>(
        hbuf, W2T, b2, x1, out, nullptr, M, D, HID, 0);
}
